// Block_Spatial_Battleneck_Share_spatial_reduction_21071109554265
// MI455X (gfx1250) — matrix operand order repaired and checked by reading, not compiled, not run on hardware
//
#include <hip/hip_runtime.h>
#include <hip/hip_bf16.h>

// ---------------------------------------------------------------------------
// Block_Spatial_Bottleneck (efficient attention + DW-conv MLP) for gfx1250.
// All 1x1 convs and the attended einsum run through v_wmma_f32_16x16x32_f16
// with f32 accumulation; BN scale/bias, ReLU, residual adds and dual
// f32/f16 stores are fused into the GEMM epilogue.
// ---------------------------------------------------------------------------

typedef __attribute__((ext_vector_type(16))) _Float16 v16h;
typedef __attribute__((ext_vector_type(8)))  float    v8f;

#define BATCH 32
#define DIM   384
#define KD    32
#define NH    8
#define MQD   256      /* NH*KD */
#define HID   1536
#define HW    1024     /* 32*32 */

// ---------------------------------------------------------------------------
// Generic WMMA GEMM:  out[z] = epilogue( W_z (MxK) @ X_z (KxN) )
//   W f32 row-major [M][K]; W_z = W + (z / wDiv) * wStride
//   X f16 row-major [K][N] per z (stride K*N)
// Block: 256 threads (8 waves). Tile BM=64, BN=128, BK=32.
// Wave w: mOff=(w&3)*16, nOff=(w>>2)*64 -> 4 WMMAs per K-step, A reused.
// ---------------------------------------------------------------------------
__global__ __launch_bounds__(256)
void gemm_wmma_f16(const float* __restrict__ W, long wStride, int wDiv,
                   const _Float16* __restrict__ X,
                   const float* __restrict__ scale,
                   const float* __restrict__ bias,
                   const float* __restrict__ R,
                   float* __restrict__ outF32,
                   _Float16* __restrict__ outF16,
                   int M, int K, int N, int relu)
{
    __shared__ __align__(16) _Float16 Alds[64][32];    // [m][k]
    __shared__ __align__(16) _Float16 Blds[128][32];   // [n][k]  (transposed)

    const int tid   = threadIdx.x;
    const int lane  = tid & 31;
    const int wave  = tid >> 5;
    const int nBase = blockIdx.x * 128;
    const int mBase = blockIdx.y * 64;
    const int z     = blockIdx.z;

    const float*    Wb = W + (long)(z / wDiv) * wStride;
    const _Float16* Xb = X + (long)z * K * N;
    const long      zo = (long)z * M * N;

    const int mOff = (wave & 3) * 16;
    const int nOff = (wave >> 2) * 64;

    v8f acc[4];
    acc[0] = (v8f){}; acc[1] = (v8f){}; acc[2] = (v8f){}; acc[3] = (v8f){};

    // ---- staging index precompute ----
    // A: 64 rows x 32 cols f32->f16; thread handles 8 consecutive cols.
    const int ar = tid >> 2;            // 0..63
    const int ac = (tid & 3) * 8;       // 0,8,16,24
    const int mg = mBase + ar;
    const int mr = (mg < M) ? mg : (M - 1);   // clamped row (always in bounds)
    const float mmask = (mg < M) ? 1.0f : 0.0f;
    // B: 32 k-rows x 128 n-cols; thread handles k-pair (kp,kp+1) x 8 n.
    const int kp = (tid >> 4) * 2;      // 0,2,..,30
    const int ns = (tid & 15) * 8;      // 0..120

    for (int k0 = 0; k0 < K; k0 += 32) {
        // ---- stage A tile ----
        {
            const float* wrow = Wb + (long)mr * K + k0 + ac;
            const float4 w0 = *(const float4*)wrow;
            const float4 w1 = *(const float4*)(wrow + 4);
            const float wv[8] = {w0.x, w0.y, w0.z, w0.w, w1.x, w1.y, w1.z, w1.w};
            union { _Float16 h[8]; uint4 v; } ah;
            #pragma unroll
            for (int j = 0; j < 8; ++j) ah.h[j] = (_Float16)(wv[j] * mmask);
            *(uint4*)&Alds[ar][ac] = ah.v;
        }
        // ---- stage B tile: two adjacent k-rows, packed to [n][k] pairs ----
        {
            const _Float16* xr0 = Xb + (long)(k0 + kp) * N + nBase + ns;
            union { uint4 v; unsigned short s[8]; } r0, r1;
            r0.v = *(const uint4*)xr0;
            r1.v = *(const uint4*)(xr0 + N);
            #pragma unroll
            for (int j = 0; j < 8; ++j) {
                const unsigned u = (unsigned)r0.s[j] | ((unsigned)r1.s[j] << 16);
                *(unsigned*)&Blds[ns + j][kp] = u;
            }
            if (k0 + 32 < K)                      // prefetch next K tile of X
                __builtin_prefetch(xr0 + 32 * (long)N, 0, 0);
        }
        __syncthreads();

        // ---- A fragment (ISA 16-bit A 16x32 layout) ----
        union { v16h v; unsigned u[8]; } af;
        {
            const int m  = mOff + (lane & 15);
            const int kh = (lane < 16) ? 0 : 8;
            #pragma unroll
            for (int v = 0; v < 8; ++v) {
                const int k = (v < 4) ? (2 * v + kh) : (16 + 2 * (v - 4) + kh);
                af.u[v] = *(const unsigned*)&Alds[m][k];
            }
        }
        // ---- 4 B fragments (ISA 16-bit B 32x16 layout) + WMMA ----
        {
            const int kb = (lane < 16) ? 0 : 8;
            #pragma unroll
            for (int sub = 0; sub < 4; ++sub) {
                union { v16h v; unsigned u[8]; } bf;
                const int n = nOff + sub * 16 + (lane & 15);
                #pragma unroll
                for (int v = 0; v < 8; ++v)
                    bf.u[v] = *(const unsigned*)&Blds[n][(v < 4) ? (kb + 2 * v) : (16 + kb + 2 * (v - 4))];
                acc[sub] = __builtin_amdgcn_wmma_f32_16x16x32_f16(
                               false, af.v, false, bf.v, (short)0, acc[sub],
                               false, false);
            }
        }
        __syncthreads();
    }

    // ---- epilogue: C/D layout -> VGPR v: M = v + 8*(lane>=16), N = lane&15
    const int mh = (lane < 16) ? 0 : 8;
    #pragma unroll
    for (int sub = 0; sub < 4; ++sub) {
        const int n = nBase + nOff + sub * 16 + (lane & 15);
        #pragma unroll
        for (int v = 0; v < 8; ++v) {
            const int m = mBase + mOff + v + mh;
            if (m < M) {
                float y = acc[sub][v];
                if (scale) y = y * scale[m] + bias[m];
                if (R)     y += R[zo + (long)m * N + n];
                if (relu)  y = fmaxf(y, 0.0f);
                const long o = zo + (long)m * N + n;
                if (outF32) outF32[o] = y;
                if (outF16) outF16[o] = (_Float16)y;
            }
        }
    }
}

// ---------------------------------------------------------------------------
__global__ __launch_bounds__(256)
void cvt_f32_f16(const float* __restrict__ in, _Float16* __restrict__ out, long n)
{
    long i = (long)blockIdx.x * 256 + threadIdx.x;
    if (i < n) out[i] = (_Float16)in[i];
}

// softmax over the 1024-wide spatial dim; one block per row
__global__ __launch_bounds__(256)
void softmax_rows(const float* __restrict__ in, float* __restrict__ out, int nc)
{
    __shared__ float red[256];
    const int tid = threadIdx.x;
    const long ro = (long)blockIdx.x * nc;
    float lmax = -3.4e38f;
    for (int c = tid; c < nc; c += 256) lmax = fmaxf(lmax, in[ro + c]);
    red[tid] = lmax; __syncthreads();
    for (int s = 128; s > 0; s >>= 1) {
        if (tid < s) red[tid] = fmaxf(red[tid], red[tid + s]);
        __syncthreads();
    }
    const float m = red[0]; __syncthreads();
    float lsum = 0.0f;
    for (int c = tid; c < nc; c += 256) lsum += __expf(in[ro + c] - m);
    red[tid] = lsum; __syncthreads();
    for (int s = 128; s > 0; s >>= 1) {
        if (tid < s) red[tid] += red[tid + s];
        __syncthreads();
    }
    const float inv = 1.0f / red[0];
    for (int c = tid; c < nc; c += 256)
        out[ro + c] = __expf(in[ro + c] - m) * inv;
}

// softmax over 32 channels per (b,h,n); writes f16 query
__global__ __launch_bounds__(256)
void q_channel_softmax(const float* __restrict__ q, _Float16* __restrict__ qs)
{
    const long t = (long)blockIdx.x * 256 + threadIdx.x;   // < B*NH*HW
    if (t >= (long)BATCH * NH * HW) return;
    const int  n  = (int)(t & (HW - 1));
    const long bh = t >> 10;
    const float* p = q + bh * KD * HW + n;
    float mx = -3.4e38f;
    #pragma unroll
    for (int d = 0; d < KD; ++d) mx = fmaxf(mx, p[d * HW]);
    float e[KD]; float s = 0.0f;
    #pragma unroll
    for (int d = 0; d < KD; ++d) { e[d] = __expf(p[d * HW] - mx); s += e[d]; }
    const float inv = 1.0f / s;
    _Float16* o = qs + bh * KD * HW + n;
    #pragma unroll
    for (int d = 0; d < KD; ++d) o[d * HW] = (_Float16)(e[d] * inv);
}

// ctxT[b][d][k] = sum_n key[b,k,n] * kv[b,d,n]   (stored transposed for GEMM A)
__global__ __launch_bounds__(256)
void context_kernel(const float* __restrict__ key, const float* __restrict__ kv,
                    float* __restrict__ ctxT)
{
    const int b = blockIdx.x;
    for (int p = threadIdx.x; p < KD * KD; p += 256) {
        const int d = p >> 5, k = p & 31;
        const float* kp = key + ((long)b * KD + k) * HW;
        const float* vp = kv  + ((long)b * KD + d) * HW;
        float acc = 0.0f;
        for (int n = 0; n < HW; ++n) acc += kp[n] * vp[n];
        ctxT[(long)b * KD * KD + d * KD + k] = acc;
    }
}

// depthwise 3x3 + bias + relu, f16 in/out
__global__ __launch_bounds__(256)
void dwconv3x3(const _Float16* __restrict__ h1, const float* __restrict__ wdw,
               const float* __restrict__ bdw, _Float16* __restrict__ h2)
{
    const long i = (long)blockIdx.x * 256 + threadIdx.x;  // < B*HID*HW
    if (i >= (long)BATCH * HID * HW) return;
    const int  xw = (int)(i & 31);
    const int  yh = (int)((i >> 5) & 31);
    const long cp = i >> 10;
    const int  ch = (int)(cp % HID);
    const _Float16* base = h1 + (cp << 10);
    const float*    wk   = wdw + ch * 9;
    float acc = bdw[ch];
    #pragma unroll
    for (int dy = -1; dy <= 1; ++dy) {
        const int yy = yh + dy;
        if (yy < 0 || yy > 31) continue;
        #pragma unroll
        for (int dx = -1; dx <= 1; ++dx) {
            const int xx = xw + dx;
            if (xx < 0 || xx > 31) continue;
            acc += wk[(dy + 1) * 3 + (dx + 1)] * (float)base[yy * 32 + xx];
        }
    }
    h2[i] = (_Float16)fmaxf(acc, 0.0f);
}

// ---------------------------------------------------------------------------
extern "C" void kernel_launch(void* const* d_in, const int* in_sizes, int n_in,
                              void* d_out, int out_size, void* d_ws, size_t ws_size,
                              hipStream_t stream)
{
    const float* x   = (const float*)d_in[0];
    const float* wq  = (const float*)d_in[2];
    const float* sq  = (const float*)d_in[3];
    const float* bq  = (const float*)d_in[4];
    const float* wkv = (const float*)d_in[5];
    const float* skv = (const float*)d_in[6];
    const float* bkv = (const float*)d_in[7];
    const float* wp  = (const float*)d_in[8];
    const float* sp  = (const float*)d_in[9];
    const float* bp  = (const float*)d_in[10];
    const float* w1  = (const float*)d_in[11];
    const float* s1  = (const float*)d_in[12];
    const float* b1  = (const float*)d_in[13];
    const float* wdw = (const float*)d_in[14];
    const float* bdw = (const float*)d_in[15];
    const float* w2  = (const float*)d_in[16];
    const float* s2  = (const float*)d_in[17];
    const float* b2  = (const float*)d_in[18];
    float* out = (float*)d_out;

    // workspace layout (bytes)
    char* base = (char*)d_ws;
    size_t off = 0;
    auto alloc = [&](size_t bytes) { void* p = base + off; off += (bytes + 255) & ~(size_t)255; return p; };
    _Float16* x_f16  = (_Float16*)alloc((size_t)BATCH * DIM * HW * 2);
    float*    q_f32  = (float*)   alloc((size_t)BATCH * MQD * HW * 4);
    _Float16* qs_f16 = (_Float16*)alloc((size_t)BATCH * MQD * HW * 2);
    float*    kv_f32 = (float*)   alloc((size_t)BATCH * KD  * HW * 4);
    float*    key_f32= (float*)   alloc((size_t)BATCH * KD  * HW * 4);
    float*    ctxT   = (float*)   alloc((size_t)BATCH * KD  * KD * 4);
    _Float16* att_f16= (_Float16*)alloc((size_t)BATCH * MQD * HW * 2);
    float*    x1_f32 = (float*)   alloc((size_t)BATCH * DIM * HW * 4);
    _Float16* x1_f16 = (_Float16*)alloc((size_t)BATCH * DIM * HW * 2);
    _Float16* h1_f16 = (_Float16*)alloc((size_t)BATCH * HID * HW * 2);
    _Float16* h2_f16 = (_Float16*)alloc((size_t)BATCH * HID * HW * 2);
    (void)ws_size; (void)in_sizes; (void)n_in; (void)out_size;

    const dim3 blk(256);

    // 0) x -> f16
    {
        const long n = (long)BATCH * DIM * HW;
        cvt_f32_f16<<<dim3((unsigned)((n + 255) / 256)), blk, 0, stream>>>(x, x_f16, n);
    }
    // 1) kv = skv*(wkv @ x)+bkv         [B,32,1024]
    gemm_wmma_f16<<<dim3(HW / 128, (KD + 63) / 64, BATCH), blk, 0, stream>>>(
        wkv, 0, 1, x_f16, skv, bkv, nullptr, kv_f32, nullptr, KD, DIM, HW, 0);
    // 2) q = sq*(wq @ x)+bq             [B,256,1024]
    gemm_wmma_f16<<<dim3(HW / 128, (MQD + 63) / 64, BATCH), blk, 0, stream>>>(
        wq, 0, 1, x_f16, sq, bq, nullptr, q_f32, nullptr, MQD, DIM, HW, 0);
    // 3) key = softmax_n(kv)
    softmax_rows<<<dim3(BATCH * KD), blk, 0, stream>>>(kv_f32, key_f32, HW);
    // 4) ctxT[b][d][k] = sum_n key*kv
    context_kernel<<<dim3(BATCH), blk, 0, stream>>>(key_f32, kv_f32, ctxT);
    // 5) query = softmax_d(q) -> f16
    q_channel_softmax<<<dim3(BATCH * NH * HW / 256), blk, 0, stream>>>(q_f32, qs_f16);
    // 6) attended = relu(ctxT @ query) per (b,h) -> f16   [B,256,1024]
    gemm_wmma_f16<<<dim3(HW / 128, 1, BATCH * NH), blk, 0, stream>>>(
        ctxT, KD * KD, NH, qs_f16, nullptr, nullptr, nullptr,
        nullptr, att_f16, KD, KD, HW, 1);
    // 7) x1 = x + sp*(wp @ att)+bp      (f32 + f16 copies)
    gemm_wmma_f16<<<dim3(HW / 128, DIM / 64, BATCH), blk, 0, stream>>>(
        wp, 0, 1, att_f16, sp, bp, x, x1_f32, x1_f16, DIM, MQD, HW, 0);
    // 8) h1 = s1*(w1 @ x1)+b1 -> f16    [B,1536,1024]
    gemm_wmma_f16<<<dim3(HW / 128, HID / 64, BATCH), blk, 0, stream>>>(
        w1, 0, 1, x1_f16, s1, b1, nullptr, nullptr, h1_f16, HID, DIM, HW, 0);
    // 9) h2 = relu(dw3x3(h1)+bdw) -> f16
    {
        const long n = (long)BATCH * HID * HW;
        dwconv3x3<<<dim3((unsigned)((n + 255) / 256)), blk, 0, stream>>>(h1_f16, wdw, bdw, h2_f16);
    }
    // 10) out = x1 + s2*(w2 @ h2)+b2    [B,384,1024] f32
    gemm_wmma_f16<<<dim3(HW / 128, DIM / 64, BATCH), blk, 0, stream>>>(
        w2, 0, 1, h2_f16, s2, b2, x1_f32, out, nullptr, DIM, HID, HW, 0);
}
